// ExphormerAttention_10445360464127
// MI455X (gfx1250) — matrix operand order repaired and checked by reading, not compiled, not run on hardware
//
#include <hip/hip_runtime.h>
#include <hip/hip_bf16.h>

// ---------------------------------------------------------------------------
// Exphormer attention, fused for MI455X (gfx1250, wave32, WMMA).
//
// Phases (all per-launch, deterministic):
//   1. convert weights fp32 -> bf16
//   2. build dst-CSR (zero / atomic count / single-block scan / atomic fill)
//   3. QKV projection via v_wmma_f32_16x16x32_bf16
//   4. fused edge projection (WMMA) + score = exp(clamp(sum(E*K*Q)/4))
//   5. atomic-free gather aggregation over CSR into d_out
// ---------------------------------------------------------------------------

typedef __attribute__((ext_vector_type(16))) __bf16 v16bf;
typedef __attribute__((ext_vector_type(8)))  float  v8f;

#define DIN  128
#define DOUT 128
#define NHEAD 8
#define DH   16

static __device__ __forceinline__ __bf16 f2bf(float f) { return (__bf16)f; }

// --------------------------- small utility kernels -------------------------

__global__ void zero_i32_kernel(int* p, int n) {
    int i = blockIdx.x * blockDim.x + threadIdx.x;
    if (i < n) p[i] = 0;
}

__global__ void cvt_w_kernel(const float* w0, const float* w1, const float* w2,
                             const float* w3, __bf16* o0, __bf16* o1,
                             __bf16* o2, __bf16* o3, int n) {
    int i = blockIdx.x * blockDim.x + threadIdx.x;
    if (i >= n) return;
    o0[i] = f2bf(w0[i]); o1[i] = f2bf(w1[i]);
    o2[i] = f2bf(w2[i]); o3[i] = f2bf(w3[i]);
}

__global__ void count_kernel(const int* ei, int ne, int* deg) {
    int e = blockIdx.x * blockDim.x + threadIdx.x;
    if (e >= ne) return;
    atomicAdd(&deg[ei[ne + e]], 1);     // dst row of edge_index
}

// single-block exclusive scan over deg[0..n) -> rowstart[0..n]
__global__ void scan_kernel(const int* deg, int* rowstart, int n) {
    __shared__ int sums[1024];
    int t = threadIdx.x;
    int chunk = (n + 1023) >> 10;
    int lo = t * chunk;
    int hi = lo + chunk; if (hi > n) hi = n; if (lo > n) lo = n;
    int s = 0;
    for (int i = lo; i < hi; ++i) s += deg[i];
    sums[t] = s;
    __syncthreads();
    if (t == 0) {
        int acc = 0;
        for (int i = 0; i < 1024; ++i) { int v = sums[i]; sums[i] = acc; acc += v; }
    }
    __syncthreads();
    int acc = sums[t];
    for (int i = lo; i < hi; ++i) { rowstart[i] = acc; acc += deg[i]; }
    if (t == 1023) rowstart[n] = acc;   // last thread's range is empty/tail -> total
}

__global__ void fill_kernel(const int* ei, int ne, int* cursor,
                            int* perm, int* csrsrc) {
    int e = blockIdx.x * blockDim.x + threadIdx.x;
    if (e >= ne) return;
    int src = ei[e];
    int dst = ei[ne + e];
    int pos = atomicAdd(&cursor[dst], 1);
    perm[pos]   = e;
    csrsrc[pos] = src;
}

// --------------------------- QKV projection (WMMA) -------------------------
// out[n,o] = sum_i x[n,i] * W[o,i] + b[o]
// blockDim = 256 (8 waves); wave w owns output col-tile w; blockIdx = row tile.
__global__ void qkv_kernel(const float* __restrict__ x,
                           const __bf16* __restrict__ wqb,
                           const __bf16* __restrict__ wkb,
                           const __bf16* __restrict__ wvb,
                           const float* __restrict__ bq,
                           const float* __restrict__ bk,
                           const float* __restrict__ bv,
                           float* __restrict__ Q, float* __restrict__ K,
                           float* __restrict__ V) {
    const int wave = threadIdx.x >> 5;
    const int lane = threadIdx.x & 31;
    const int rowbase = blockIdx.x * 16;
    const int colbase = wave * 16;
    const int m    = lane & 15;
    const int half = lane >> 4;

    v8f accq = {}, acck = {}, accv = {};

    #pragma unroll
    for (int ks = 0; ks < 4; ++ks) {
        const int kb = ks * 32;
        // A tile: 16x32 bf16, contiguous k = 16*half + s (same order as B)
        v16bf a;
        const float* xrow = x + (size_t)(rowbase + m) * DIN + kb;
        #pragma unroll
        for (int j = 0; j < 8; ++j) {
            int k = half * 16 + j * 2;
            float2 f = *(const float2*)(xrow + k);
            a[2 * j]     = f2bf(f.x);
            a[2 * j + 1] = f2bf(f.y);
        }
        // B tile: 16 contiguous bf16 of weight row (n = lane&15, K block by half)
        const size_t wo = (size_t)(colbase + m) * DIN + kb + half * 16;
        v16bf bqm = *(const v16bf*)(wqb + wo);
        v16bf bkm = *(const v16bf*)(wkb + wo);
        v16bf bvm = *(const v16bf*)(wvb + wo);
        accq = __builtin_amdgcn_wmma_f32_16x16x32_bf16(false, a, false, bqm, (short)0, accq, false, false);
        acck = __builtin_amdgcn_wmma_f32_16x16x32_bf16(false, a, false, bkm, (short)0, acck, false, false);
        accv = __builtin_amdgcn_wmma_f32_16x16x32_bf16(false, a, false, bvm, (short)0, accv, false, false);
    }

    const int n = lane & 15;
    const float fq = bq[colbase + n], fk = bk[colbase + n], fv = bv[colbase + n];
    #pragma unroll
    for (int r = 0; r < 8; ++r) {
        const size_t row = rowbase + half * 8 + r;
        Q[row * DOUT + colbase + n] = accq[r] + fq;
        K[row * DOUT + colbase + n] = acck[r] + fk;
        V[row * DOUT + colbase + n] = accv[r] + fv;
    }
}

// ------------------- fused edge projection + attention score ---------------
// One wave per 16-edge tile. E tile lives only in accumulators; head h == col
// tile h (DH==16). score[e*8+h] = exp(clamp(sum_d E*K[src]*Q[dst]/4, -5, 5)).
__global__ void edge_kernel(const float* __restrict__ ea,
                            const __bf16* __restrict__ web,
                            const float* __restrict__ be,
                            const int* __restrict__ ei,
                            const float* __restrict__ Q,
                            const float* __restrict__ K,
                            float* __restrict__ score, int ne) {
    const int wave = threadIdx.x >> 5;
    const int lane = threadIdx.x & 31;
    const int tile = blockIdx.x * 4 + wave;
    const int ebase = tile * 16;
    const int m16  = lane & 15;
    const int half = lane >> 4;

    // prefetch next tile's edge_attr rows (HBM streaming)
    if ((ebase + 16) * (size_t)DIN < (size_t)ne * DIN)
        __builtin_prefetch(ea + (size_t)(ebase + 16) * DIN + m16 * DIN, 0, 1);

    v8f acc[8] = {};
    #pragma unroll
    for (int ks = 0; ks < 4; ++ks) {
        const int kb = ks * 32;
        v16bf a;
        const float* arow = ea + (size_t)(ebase + m16) * DIN + kb;
        #pragma unroll
        for (int j = 0; j < 8; ++j) {
            int k = half * 16 + j * 2;
            float2 f = *(const float2*)(arow + k);
            a[2 * j]     = f2bf(f.x);
            a[2 * j + 1] = f2bf(f.y);
        }
        const int kbb = kb + half * 16;
        #pragma unroll
        for (int c = 0; c < 8; ++c) {
            v16bf b = *(const v16bf*)(web + (size_t)(c * 16 + m16) * DIN + kbb);
            acc[c] = __builtin_amdgcn_wmma_f32_16x16x32_bf16(false, a, false, b, (short)0, acc[c], false, false);
        }
    }

    // gather edge endpoints once (uniform per 16-lane half for each r)
    int srcs[8], dsts[8];
    #pragma unroll
    for (int r = 0; r < 8; ++r) {
        int e = ebase + half * 8 + r;
        srcs[r] = ei[e];
        dsts[r] = ei[ne + e];
    }

    const float scale = 0.25f;          // 1/sqrt(DH), DH = 16
    const int d = lane & 15;            // D-matrix N index = head dim
    #pragma unroll
    for (int h = 0; h < 8; ++h) {
        const float beh = be[h * 16 + d];
        #pragma unroll
        for (int r = 0; r < 8; ++r) {
            const float kv = K[(size_t)srcs[r] * DOUT + h * 16 + d];
            const float qv = Q[(size_t)dsts[r] * DOUT + h * 16 + d];
            float prod = (acc[h][r] + beh) * kv * qv * scale;
            // reduce the 16 lanes of this half (rows m=half*8+r stay separate)
            prod += __shfl_xor(prod, 1, 32);
            prod += __shfl_xor(prod, 2, 32);
            prod += __shfl_xor(prod, 4, 32);
            prod += __shfl_xor(prod, 8, 32);
            if (d == 0) {
                float s = fminf(fmaxf(prod, -5.0f), 5.0f);
                score[(size_t)(ebase + half * 8 + r) * NHEAD + h] = __expf(s);
            }
        }
    }
}

// --------------------- atomic-free CSR aggregation -------------------------
// One wave per dst node; lane owns 4 contiguous dims (within one head).
__global__ void agg_kernel(const float* __restrict__ V,
                           const float* __restrict__ score,
                           const int* __restrict__ rowstart,
                           const int* __restrict__ perm,
                           const int* __restrict__ csrsrc,
                           float* __restrict__ out, int n) {
    const int node = blockIdx.x * 8 + (threadIdx.x >> 5);
    const int lane = threadIdx.x & 31;
    if (node >= n) return;
    const int beg = rowstart[node];
    const int end = rowstart[node + 1];
    const int h = lane >> 2;            // head of dims [lane*4, lane*4+4)
    float4 acc = {0.f, 0.f, 0.f, 0.f};
    for (int s = beg; s < end; ++s) {
        const int e   = perm[s];
        const int src = csrsrc[s];
        const float sc = score[(size_t)e * NHEAD + h];
        const float4 v = *(const float4*)(V + (size_t)src * DOUT + lane * 4);
        acc.x += v.x * sc; acc.y += v.y * sc;
        acc.z += v.z * sc; acc.w += v.w * sc;
    }
    *(float4*)(out + (size_t)node * DOUT + lane * 4) = acc;
}

// ---------------------------------------------------------------------------

extern "C" void kernel_launch(void* const* d_in, const int* in_sizes, int n_in,
                              void* d_out, int out_size, void* d_ws, size_t ws_size,
                              hipStream_t stream) {
    const float* x  = (const float*)d_in[0];
    const float* ea = (const float*)d_in[1];
    const int*   ei = (const int*)d_in[2];
    const int N  = in_sizes[3];          // batch_vec length = num nodes
    const int NE = in_sizes[2] / 2;
    const float* Wq = (const float*)d_in[4];  const float* bq = (const float*)d_in[5];
    const float* Wk = (const float*)d_in[6];  const float* bk = (const float*)d_in[7];
    const float* We = (const float*)d_in[8];  const float* be = (const float*)d_in[9];
    const float* Wv = (const float*)d_in[10]; const float* bv = (const float*)d_in[11];
    float* out = (float*)d_out;

    // workspace carve-up (256B aligned)
    char* ws = (char*)d_ws;
    size_t off = 0;
    auto carve = [&](size_t bytes) {
        void* p = ws + off;
        off += (bytes + 255) & ~(size_t)255;
        return p;
    };
    float*  Qbuf   = (float*)carve((size_t)N * DOUT * 4);
    float*  Kbuf   = (float*)carve((size_t)N * DOUT * 4);
    float*  Vbuf   = (float*)carve((size_t)N * DOUT * 4);
    float*  scoreb = (float*)carve((size_t)NE * NHEAD * 4);
    int*    deg    = (int*)carve((size_t)N * 4);
    int*    rowst  = (int*)carve((size_t)(N + 1) * 4);
    int*    cursor = (int*)carve((size_t)(N + 1) * 4);
    int*    perm   = (int*)carve((size_t)NE * 4);
    int*    csrsrc = (int*)carve((size_t)NE * 4);
    __bf16* wqb    = (__bf16*)carve((size_t)DOUT * DIN * 2);
    __bf16* wkb    = (__bf16*)carve((size_t)DOUT * DIN * 2);
    __bf16* wvb    = (__bf16*)carve((size_t)DOUT * DIN * 2);
    __bf16* web    = (__bf16*)carve((size_t)DOUT * DIN * 2);
    (void)ws_size; (void)n_in; (void)out_size;

    const int WN = DOUT * DIN;

    // 1. weights -> bf16
    cvt_w_kernel<<<(WN + 255) / 256, 256, 0, stream>>>(Wq, Wk, Wv, We,
                                                       wqb, wkb, wvb, web, WN);
    // 2. CSR build
    zero_i32_kernel<<<(N + 255) / 256, 256, 0, stream>>>(deg, N);
    count_kernel<<<(NE + 255) / 256, 256, 0, stream>>>(ei, NE, deg);
    scan_kernel<<<1, 1024, 0, stream>>>(deg, rowst, N);
    hipMemcpyAsync(cursor, rowst, (size_t)N * 4, hipMemcpyDeviceToDevice, stream);
    fill_kernel<<<(NE + 255) / 256, 256, 0, stream>>>(ei, NE, cursor, perm, csrsrc);

    // 3. QKV projection (N multiple of 16: 50000/16 = 3125 row tiles)
    qkv_kernel<<<N / 16, 256, 0, stream>>>(x, wqb, wkb, wvb, bq, bk, bv,
                                           Qbuf, Kbuf, Vbuf);

    // 4. fused edge projection + scores (NE multiple of 64: 12500 blocks x 4 waves)
    edge_kernel<<<NE / 64, 128, 0, stream>>>(ea, web, be, ei, Qbuf, Kbuf,
                                             scoreb, NE);

    // 5. aggregation, one wave per node (N multiple of 8)
    agg_kernel<<<(N + 7) / 8, 256, 0, stream>>>(Vbuf, scoreb, rowst, perm,
                                                csrsrc, out, N);
}
